// GATEncoder_9320079033061
// MI455X (gfx1250) — matrix operand order repaired and checked by reading, not compiled, not run on hardware
//
#include <hip/hip_runtime.h>
#include <hip/hip_bf16.h>

typedef __attribute__((ext_vector_type(16))) _Float16 v16h;
typedef __attribute__((ext_vector_type(8)))  _Float16 v8h;
typedef __attribute__((ext_vector_type(8)))  float    v8f;

#define TPB 256
static inline unsigned ceil_div(long long n, int d) { return (unsigned)((n + d - 1) / d); }

// ---- float <-> order-preserving uint (for atomic segment-max) ----
__device__ __forceinline__ unsigned f2ord(float f) {
  unsigned u = __float_as_uint(f);
  return (u & 0x80000000u) ? ~u : (u | 0x80000000u);
}
__device__ __forceinline__ float ord2f(unsigned u) {
  u = (u & 0x80000000u) ? (u ^ 0x80000000u) : ~u;
  return __uint_as_float(u);
}
#define ORD_NEG_INF 0x007FFFFFu   // f2ord(-inf)

__device__ __forceinline__ float lrelu(float x) { return x > 0.f ? x : 0.2f * x; }

// ------------------------- small utility kernels -------------------------
__global__ void k_fill_f32(float* p, int n, float v) {
  int i = blockIdx.x * blockDim.x + threadIdx.x; if (i < n) p[i] = v;
}
__global__ void k_fill_u32(unsigned* p, int n, unsigned v) {
  int i = blockIdx.x * blockDim.x + threadIdx.x; if (i < n) p[i] = v;
}
// broadcast bias into an accumulator buffer: p[i] = b[i & mask]
__global__ void k_bcast_bias(float* p, const float* __restrict__ b, int n, int mask) {
  int i = blockIdx.x * blockDim.x + threadIdx.x; if (i < n) p[i] = b[i & mask];
}
__global__ void k_f32_to_f16(const float* __restrict__ in, _Float16* __restrict__ out, int n) {
  int i = blockIdx.x * blockDim.x + threadIdx.x; if (i < n) out[i] = (_Float16)in[i];
}
// W[K][N] (row major f32) -> Bt[N][K] (row major f16)
__global__ void k_transpose_f16(const float* __restrict__ W, _Float16* __restrict__ Bt, int K, int N) {
  int i = blockIdx.x * blockDim.x + threadIdx.x;
  if (i < K * N) { int k = i / N, n = i - k * N; Bt[(size_t)n * K + k] = (_Float16)W[i]; }
}

// ------------------- self-loop attr (fill_value='mean') -------------------
__global__ void k_deg_sum(const int* __restrict__ dst, const float* __restrict__ ea,
                          float* deg, float* asum, int E) {
  int e = blockIdx.x * blockDim.x + threadIdx.x;
  if (e >= E) return;
  int d = dst[e];
  atomicAdd(&deg[d], 1.f);
  atomicAdd(&asum[2 * d + 0], ea[2 * e + 0]);
  atomicAdd(&asum[2 * d + 1], ea[2 * e + 1]);
}
__global__ void k_loop_attr(const float* __restrict__ deg, const float* __restrict__ asum,
                            float* lattr, int N) {
  int n = blockIdx.x * blockDim.x + threadIdx.x;
  if (n >= N) return;
  float dd = fmaxf(deg[n], 1.f);
  lattr[2 * n + 0] = asum[2 * n + 0] / dd;
  lattr[2 * n + 1] = asum[2 * n + 1] / dd;
}

// ----------------------------- WMMA GEMM -----------------------------
// C[M][N] f32 = A[M][K] f16 (row major) x Bt[N][K] f16 (B pre-transposed).
// One wave computes a 16x64 tile of C (A fragment reused across 4 WMMAs).
// M % 16 == 0, N % 64 == 0, K % 32 == 0.
__global__ __launch_bounds__(256, 1)
void k_wmma_gemm_n64(const _Float16* __restrict__ A, const _Float16* __restrict__ Bt,
                     float* __restrict__ C, int M, int N, int K) {
  const int lane = threadIdx.x & 31;
  const int wave = threadIdx.x >> 5;
  const int half = lane >> 4;      // 0: lanes 0-15, 1: lanes 16-31
  const int l16  = lane & 15;
  const int tiles_n4 = N >> 6;
  const int tiles_m  = M >> 4;
  const int tile = blockIdx.x * 8 + wave;
  if (tile >= tiles_m * tiles_n4) return;
  const int tm  = tile / tiles_n4;
  const int tn4 = tile - tm * tiles_n4;

  const _Float16* arow = A  + (size_t)(tm * 16 + l16) * K;
  const _Float16* pb0  = Bt + (size_t)(tn4 * 64 +  0 + l16) * K;
  const _Float16* pb1  = Bt + (size_t)(tn4 * 64 + 16 + l16) * K;
  const _Float16* pb2  = Bt + (size_t)(tn4 * 64 + 32 + l16) * K;
  const _Float16* pb3  = Bt + (size_t)(tn4 * 64 + 48 + l16) * K;

  v8f c0 = {}, c1 = {}, c2 = {}, c3 = {};
  for (int kk = 0; kk < K; kk += 32) {
    // A 16x32 f16 fragment: lanes 0-15 hold K[kk+0..15],
    // lanes 16-31 hold K[kk+16..31]  (contiguous, matching B).
    const int ka = kk + half * 16;
    v8h alo = *(const v8h*)(arow + ka);
    v8h ahi = *(const v8h*)(arow + ka + 8);
    v16h a;
#pragma unroll
    for (int i = 0; i < 8; ++i) { a[i] = alo[i]; a[i + 8] = ahi[i]; }
    // B 32x16 f16 fragment: lane = column, lanes 0-15 K[kk+0..15],
    // lanes 16-31 K[kk+16..31] -> contiguous in Bt rows.
    const int kb = kk + half * 16;
    v16h vb0 = *(const v16h*)(pb0 + kb);
    v16h vb1 = *(const v16h*)(pb1 + kb);
    v16h vb2 = *(const v16h*)(pb2 + kb);
    v16h vb3 = *(const v16h*)(pb3 + kb);
    c0 = __builtin_amdgcn_wmma_f32_16x16x32_f16(false, a, false, vb0, (short)0, c0, false, false);
    c1 = __builtin_amdgcn_wmma_f32_16x16x32_f16(false, a, false, vb1, (short)0, c1, false, false);
    c2 = __builtin_amdgcn_wmma_f32_16x16x32_f16(false, a, false, vb2, (short)0, c2, false, false);
    c3 = __builtin_amdgcn_wmma_f32_16x16x32_f16(false, a, false, vb3, (short)0, c3, false, false);
  }
  // C layout: VGPR r holds row (8*half + r), column = l16.
  const int rowbase = tm * 16 + half * 8;
  float* crow = C + (size_t)rowbase * N + tn4 * 64 + l16;
#pragma unroll
  for (int r = 0; r < 8; ++r) {
    crow[(size_t)r * N +  0] = c0[r];
    crow[(size_t)r * N + 16] = c1[r];
    crow[(size_t)r * N + 32] = c2[r];
    crow[(size_t)r * N + 48] = c3[r];
  }
}

// ---------------------- attention scalar projections ----------------------
// layer1: per (node, head) dot over 64 dims
__global__ void k_alpha1(const float* __restrict__ h1, const float* __restrict__ a_src,
                         const float* __restrict__ a_dst, float* as1, float* ad1, int N) {
  int t = blockIdx.x * blockDim.x + threadIdx.x;
  if (t >= N * 4) return;
  int n = t >> 2, h = t & 3;
  const float* hp = h1 + (size_t)n * 256 + h * 64;
  const float* sp = a_src + h * 64;
  const float* dp = a_dst + h * 64;
  float s1 = 0.f, s2 = 0.f;
  for (int d = 0; d < 64; ++d) { float v = hp[d]; s1 += v * sp[d]; s2 += v * dp[d]; }
  as1[t] = s1; ad1[t] = s2;
}
// layer2: per node dot over 128 dims (1 head)
__global__ void k_alpha2(const float* __restrict__ h2, const float* __restrict__ a_src,
                         const float* __restrict__ a_dst, float* as2, float* ad2, int N) {
  int n = blockIdx.x * blockDim.x + threadIdx.x;
  if (n >= N) return;
  const float* hp = h2 + (size_t)n * 128;
  float s1 = 0.f, s2 = 0.f;
  for (int d = 0; d < 128; ++d) { float v = hp[d]; s1 += v * a_src[d]; s2 += v * a_dst[d]; }
  as2[n] = s1; ad2[n] = s2;
}
// fold (edge_attr @ We) . a_edge  into a 2-vector per head: alpha_edge = attr . ve
__global__ void k_ve1(const float* __restrict__ We1, const float* __restrict__ aedge1, float* ve1) {
  int t = threadIdx.x; if (t >= 8) return;
  int h = t >> 1, k = t & 1;
  float s = 0.f;
  for (int d = 0; d < 64; ++d) s += We1[k * 256 + h * 64 + d] * aedge1[h * 64 + d];
  ve1[h * 2 + k] = s;
}
__global__ void k_ve2(const float* __restrict__ We2, const float* __restrict__ aedge2, float* ve2) {
  int t = threadIdx.x; if (t >= 2) return;
  float s = 0.f;
  for (int d = 0; d < 128; ++d) s += We2[t * 128 + d] * aedge2[d];
  ve2[t] = s;
}

// -------------------------- layer-1 edge passes --------------------------
__global__ void k_edge_max1(const int* __restrict__ src, const int* __restrict__ dst,
                            const float* __restrict__ ea, const float* __restrict__ lattr,
                            const float* __restrict__ as1, const float* __restrict__ ad1,
                            const float* __restrict__ ve1, unsigned* amax, int E, int ET) {
  int e = blockIdx.x * blockDim.x + threadIdx.x;
  if (e >= ET) return;
  int s, d; float a0, a1;
  if (e < E) { s = src[e]; d = dst[e]; a0 = ea[2 * e]; a1 = ea[2 * e + 1]; }
  else       { s = d = e - E; a0 = lattr[2 * s]; a1 = lattr[2 * s + 1]; }
#pragma unroll
  for (int h = 0; h < 4; ++h) {
    float al = lrelu(as1[s * 4 + h] + ad1[d * 4 + h] + a0 * ve1[2 * h] + a1 * ve1[2 * h + 1]);
    atomicMax(&amax[d * 4 + h], f2ord(al));
  }
}
__global__ void k_edge_den1(const int* __restrict__ src, const int* __restrict__ dst,
                            const float* __restrict__ ea, const float* __restrict__ lattr,
                            const float* __restrict__ as1, const float* __restrict__ ad1,
                            const float* __restrict__ ve1, const unsigned* __restrict__ amax,
                            float* ex1, float* den1, int E, int ET) {
  int e = blockIdx.x * blockDim.x + threadIdx.x;
  if (e >= ET) return;
  int s, d; float a0, a1;
  if (e < E) { s = src[e]; d = dst[e]; a0 = ea[2 * e]; a1 = ea[2 * e + 1]; }
  else       { s = d = e - E; a0 = lattr[2 * s]; a1 = lattr[2 * s + 1]; }
#pragma unroll
  for (int h = 0; h < 4; ++h) {
    float al = lrelu(as1[s * 4 + h] + ad1[d * 4 + h] + a0 * ve1[2 * h] + a1 * ve1[2 * h + 1]);
    float ex = expf(al - ord2f(amax[d * 4 + h]));
    ex1[e * 4 + h] = ex;
    atomicAdd(&den1[d * 4 + h], ex);
  }
}
__global__ void k_edge_agg1(const int* __restrict__ src, const int* __restrict__ dst,
                            const float* __restrict__ ex1, const float* __restrict__ den1,
                            const float* __restrict__ h1, float* agg1, int E, int total) {
  int t = blockIdx.x * blockDim.x + threadIdx.x;
  if (t >= total) return;                       // total = ET*64
  int e = t >> 6, dd = t & 63;
  int s, d;
  if (e < E) { s = src[e]; d = dst[e]; } else { s = d = e - E; }
#pragma unroll
  for (int h = 0; h < 4; ++h) {
    float coef = ex1[e * 4 + h] / (den1[d * 4 + h] + 1e-16f);
    atomicAdd(&agg1[(size_t)d * 256 + h * 64 + dd],
              h1[(size_t)s * 256 + h * 64 + dd] * coef);
  }
}
// ELU + convert to f16 for GEMM2 A-matrix (agg1 was pre-seeded with bias b1)
__global__ void k_elu_f16(const float* __restrict__ agg1, _Float16* __restrict__ h1h, int n) {
  int i = blockIdx.x * blockDim.x + threadIdx.x;
  if (i >= n) return;
  float v = agg1[i];
  v = v > 0.f ? v : (expf(v) - 1.f);
  h1h[i] = (_Float16)v;
}

// -------------------------- layer-2 edge passes --------------------------
__global__ void k_edge_max2(const int* __restrict__ src, const int* __restrict__ dst,
                            const float* __restrict__ ea, const float* __restrict__ lattr,
                            const float* __restrict__ as2, const float* __restrict__ ad2,
                            const float* __restrict__ ve2, unsigned* amax, int E, int ET) {
  int e = blockIdx.x * blockDim.x + threadIdx.x;
  if (e >= ET) return;
  int s, d; float a0, a1;
  if (e < E) { s = src[e]; d = dst[e]; a0 = ea[2 * e]; a1 = ea[2 * e + 1]; }
  else       { s = d = e - E; a0 = lattr[2 * s]; a1 = lattr[2 * s + 1]; }
  float al = lrelu(as2[s] + ad2[d] + a0 * ve2[0] + a1 * ve2[1]);
  atomicMax(&amax[d], f2ord(al));
}
__global__ void k_edge_den2(const int* __restrict__ src, const int* __restrict__ dst,
                            const float* __restrict__ ea, const float* __restrict__ lattr,
                            const float* __restrict__ as2, const float* __restrict__ ad2,
                            const float* __restrict__ ve2, const unsigned* __restrict__ amax,
                            float* ex2, float* den2, int E, int ET) {
  int e = blockIdx.x * blockDim.x + threadIdx.x;
  if (e >= ET) return;
  int s, d; float a0, a1;
  if (e < E) { s = src[e]; d = dst[e]; a0 = ea[2 * e]; a1 = ea[2 * e + 1]; }
  else       { s = d = e - E; a0 = lattr[2 * s]; a1 = lattr[2 * s + 1]; }
  float al = lrelu(as2[s] + ad2[d] + a0 * ve2[0] + a1 * ve2[1]);
  float ex = expf(al - ord2f(amax[d]));
  ex2[e] = ex;
  atomicAdd(&den2[d], ex);
}
__global__ void k_edge_agg2(const int* __restrict__ src, const int* __restrict__ dst,
                            const float* __restrict__ ex2, const float* __restrict__ den2,
                            const float* __restrict__ h2, float* out, int E, int total) {
  int t = blockIdx.x * blockDim.x + threadIdx.x;
  if (t >= total) return;                       // total = ET*128
  int e = t >> 7, dd = t & 127;
  int s, d;
  if (e < E) { s = src[e]; d = dst[e]; } else { s = d = e - E; }
  float coef = ex2[e] / (den2[d] + 1e-16f);
  atomicAdd(&out[(size_t)d * 128 + dd], h2[(size_t)s * 128 + dd] * coef);
}

// ------------------------------ launcher ------------------------------
extern "C" void kernel_launch(void* const* d_in, const int* in_sizes, int n_in,
                              void* d_out, int out_size, void* d_ws, size_t ws_size,
                              hipStream_t stream) {
  const int N = 50000, E = 800000, ET = E + N;
  const float* x      = (const float*)d_in[0];
  const int*   ei     = (const int*)d_in[1];
  const float* ea     = (const float*)d_in[2];
  const float* W1     = (const float*)d_in[3];
  const float* We1    = (const float*)d_in[4];
  const float* asrc1  = (const float*)d_in[5];
  const float* adst1  = (const float*)d_in[6];
  const float* aedge1 = (const float*)d_in[7];
  const float* b1     = (const float*)d_in[8];
  const float* W2     = (const float*)d_in[9];
  const float* We2    = (const float*)d_in[10];
  const float* asrc2  = (const float*)d_in[11];
  const float* adst2  = (const float*)d_in[12];
  const float* aedge2 = (const float*)d_in[13];
  const float* b2     = (const float*)d_in[14];
  float* out = (float*)d_out;
  const int* src = ei;          // edge_index[0]
  const int* dst = ei + E;      // edge_index[1]

  char* ws = (char*)d_ws;
  size_t off = 0;
  auto alloc = [&](size_t bytes) -> void* {
    void* p = ws + off;
    off = (off + bytes + 255) & ~(size_t)255;
    return p;
  };
  float*    deg   = (float*)alloc((size_t)N * 4);
  float*    asum  = (float*)alloc((size_t)N * 2 * 4);
  float*    lattr = (float*)alloc((size_t)N * 2 * 4);
  _Float16* xh    = (_Float16*)alloc((size_t)N * 128 * 2);
  _Float16* bt1   = (_Float16*)alloc((size_t)256 * 128 * 2);
  float*    h1    = (float*)alloc((size_t)N * 256 * 4);
  float*    as1   = (float*)alloc((size_t)N * 4 * 4);
  float*    ad1   = (float*)alloc((size_t)N * 4 * 4);
  float*    ve1   = (float*)alloc(8 * 4);
  unsigned* amax1 = (unsigned*)alloc((size_t)N * 4 * 4);
  float*    den1  = (float*)alloc((size_t)N * 4 * 4);
  float*    ex1   = (float*)alloc((size_t)ET * 4 * 4);
  float*    agg1  = (float*)alloc((size_t)N * 256 * 4);
  _Float16* h1h   = (_Float16*)alloc((size_t)N * 256 * 2);
  _Float16* bt2   = (_Float16*)alloc((size_t)128 * 256 * 2);
  float*    h2    = (float*)alloc((size_t)N * 128 * 4);
  float*    as2   = (float*)alloc((size_t)N * 4);
  float*    ad2   = (float*)alloc((size_t)N * 4);
  float*    ve2   = (float*)alloc(2 * 4);
  unsigned* amax2 = (unsigned*)alloc((size_t)N * 4);
  float*    den2  = (float*)alloc((size_t)N * 4);
  float*    ex2   = (float*)alloc((size_t)ET * 4);
  if (off > ws_size) return;  // workspace too small: nothing safe to do

#define LAUNCH(kern, n, ...) kern<<<dim3(ceil_div((n), TPB)), dim3(TPB), 0, stream>>>(__VA_ARGS__)

  // --- init accumulators (re-done every call; deterministic) ---
  LAUNCH(k_fill_f32, N,          deg,  N, 0.f);
  LAUNCH(k_fill_f32, 2 * N,      asum, 2 * N, 0.f);
  LAUNCH(k_fill_f32, 4 * N,      den1, 4 * N, 0.f);
  LAUNCH(k_fill_f32, N,          den2, N, 0.f);
  LAUNCH(k_fill_u32, 4 * N,      amax1, 4 * N, ORD_NEG_INF);
  LAUNCH(k_fill_u32, N,          amax2, N, ORD_NEG_INF);
  LAUNCH(k_bcast_bias, N * 256,  agg1, b1, N * 256, 255);
  LAUNCH(k_bcast_bias, N * 128,  out,  b2, N * 128, 127);

  // --- self-loop attr = mean of incoming edge_attr ---
  LAUNCH(k_deg_sum, E, dst, ea, deg, asum, E);
  LAUNCH(k_loop_attr, N, deg, asum, lattr, N);

  // --- layer 1: h1 = x @ W1 via WMMA (f16 in, f32 accum) ---
  LAUNCH(k_f32_to_f16, N * 128, x, xh, N * 128);
  LAUNCH(k_transpose_f16, 128 * 256, W1, bt1, 128, 256);
  {
    int tiles = (N / 16) * (256 / 64);           // 12500 wave-tiles, 8 waves/block
    k_wmma_gemm_n64<<<dim3((tiles + 7) / 8), dim3(256), 0, stream>>>(xh, bt1, h1, N, 256, 128);
  }
  LAUNCH(k_alpha1, N * 4, h1, asrc1, adst1, as1, ad1, N);
  k_ve1<<<dim3(1), dim3(8), 0, stream>>>(We1, aedge1, ve1);

  LAUNCH(k_edge_max1, ET, src, dst, ea, lattr, as1, ad1, ve1, amax1, E, ET);
  LAUNCH(k_edge_den1, ET, src, dst, ea, lattr, as1, ad1, ve1, amax1, ex1, den1, E, ET);
  LAUNCH(k_edge_agg1, ET * 64, src, dst, ex1, den1, h1, agg1, E, ET * 64);

  // --- ELU + f16 convert, then layer 2 GEMM: h2 = elu(h1agg) @ W2 ---
  LAUNCH(k_elu_f16, N * 256, agg1, h1h, N * 256);
  LAUNCH(k_transpose_f16, 256 * 128, W2, bt2, 256, 128);
  {
    int tiles = (N / 16) * (128 / 64);           // 6250 wave-tiles
    k_wmma_gemm_n64<<<dim3((tiles + 7) / 8), dim3(256), 0, stream>>>(h1h, bt2, h2, N, 128, 256);
  }
  LAUNCH(k_alpha2, N, h2, asrc2, adst2, as2, ad2, N);
  k_ve2<<<dim3(1), dim3(2), 0, stream>>>(We2, aedge2, ve2);

  LAUNCH(k_edge_max2, ET, src, dst, ea, lattr, as2, ad2, ve2, amax2, E, ET);
  LAUNCH(k_edge_den2, ET, src, dst, ea, lattr, as2, ad2, ve2, amax2, ex2, den2, E, ET);
  LAUNCH(k_edge_agg2, ET * 128, src, dst, ex2, den2, h2, out, E, ET * 128);
#undef LAUNCH
}
